// MFCCRNN_45904610459658
// MI455X (gfx1250) — matrix operand order repaired and checked by reading, not compiled, not run on hardware
//
#include <hip/hip_runtime.h>

// MI455X / gfx1250, wave32. Fused persistent BiGRU scan + WMMA bf16.

typedef __attribute__((ext_vector_type(16))) __bf16 v16bf;
typedef __attribute__((ext_vector_type(8)))  float  v8f;

#define T_STEPS 1000
#define BATCH   128
#define HDIM    128

__device__ inline unsigned short f2bf(float f) {
    union { float f; unsigned u; } v; v.f = f;
    unsigned r = v.u + 0x7FFFu + ((v.u >> 16) & 1u);   // round-to-nearest-even
    return (unsigned short)(r >> 16);
}
__device__ inline float sigmoidf_(float x) { return 1.0f / (1.0f + __expf(-x)); }
__device__ inline float tanhf_(float x)    { return 1.0f - 2.0f / (1.0f + __expf(2.0f * x)); }

union Frag { v16bf v; unsigned short us[16]; unsigned u32[8]; };

// One workgroup = 32 waves (1024 thr) = one direction x one 16-row batch slice.
// Wave w owns gate-column tile w (N = 512 = 4 groups {r,z,xn,hn} x 128 units).
// B fragments (weights) resident in VGPRs for the whole T-loop.
// A = [x_t | h] bf16 in LDS, rebuilt per step; h state f32 in LDS.
// x_{t+1} staging is software-pipelined through registers to hide L2 latency.
template <int KXPAD, int KTOT, int KFRAGS, int MODE>
__global__ __launch_bounds__(1024) void bigru_scan(
    const void* __restrict__ xin_,
    const float* __restrict__ w_ih, const float* __restrict__ w_hh,
    const float* __restrict__ b_ih, const float* __restrict__ b_hh,
    int din,
    unsigned short* __restrict__ out_bf16,  // MODE 0: [B,T,2H] bf16
    float* __restrict__ pooled)             // MODE 1: [B,2H]  f32 (time mean)
{
    __shared__ unsigned short Ash[16 * KTOT];   // [x_t | h] bf16, row-major K
    __shared__ float Hg[16 * 512];              // gate pre-activations
    __shared__ float Hs[16 * HDIM];             // f32 hidden state

    const int tid  = threadIdx.x;
    const int lane = tid & 31;
    const int wv   = tid >> 5;                  // 0..31 : column tile
    const int half = (lane >> 4) & 1;
    const int ln   = lane & 15;

    const int dir = blockIdx.x >> 3;            // 0 fwd, 1 bwd
    const int b0  = (blockIdx.x & 7) * 16;      // batch slice base

    const float* Wih = w_ih + (size_t)dir * 384 * din;
    const float* Whh = w_hh + (size_t)dir * 384 * HDIM;
    const float* Bih = b_ih + dir * 384;
    const float* Bhh = b_hh + dir * 384;

    const int cg   = wv >> 3;                   // 0=r, 1=z, 2=xn, 3=hn
    const int ucol = (wv & 7) * 16 + ln;        // hidden unit 0..127 for this lane
    const int wrow = (cg == 0) ? ucol : (cg == 1) ? (128 + ucol) : (256 + ucol);

    // ---- load resident B fragments (bf16), split x-part / h-part per group ----
    Frag bfrag[KFRAGS];
#pragma unroll
    for (int kf = 0; kf < KFRAGS; ++kf) {
#pragma unroll
        for (int i = 0; i < 8; ++i) {
#pragma unroll
            for (int j = 0; j < 2; ++j) {
                int k = 32 * kf + 8 * half + ((i < 4) ? (2 * i) : (2 * i + 8)) + j;   // B: lanes0-15 K {0..7,16..23}, lanes16-31 {8..15,24..31}
                float w;
                if (k < KXPAD) {                            // x region
                    w = (cg == 3) ? 0.0f : ((k < din) ? Wih[(size_t)wrow * din + k] : 0.0f);
                } else {                                    // h region
                    int kk = k - KXPAD;
                    w = (cg == 2) ? 0.0f : Whh[(size_t)wrow * HDIM + kk];
                }
                bfrag[kf].us[2 * i + j] = f2bf(w);
            }
        }
    }
    const float bias = (cg == 0) ? (Bih[ucol] + Bhh[ucol])
                     : (cg == 1) ? (Bih[128 + ucol] + Bhh[128 + ucol])
                     : (cg == 2) ? Bih[256 + ucol]
                                 : Bhh[256 + ucol];

    // ---- init LDS state ----
    for (int e = tid; e < 16 * KTOT; e += 1024) Ash[e] = 0;
    for (int e = tid; e < 16 * HDIM; e += 1024) Hs[e]  = 0.0f;

    float pacc0 = 0.0f, pacc1 = 0.0f;

    const float*          xf = (const float*)xin_;           // MODE 0: [B,T,din] f32
    const unsigned short* xb = (const unsigned short*)xin_;  // MODE 1: [B,T,256] bf16

    // per-thread staging slot: MODE0 -> 1 elem, MODE1 -> 4 consecutive bf16 (8B)
    const int sb = tid >> 6;                                  // batch row 0..15
    const int sk = (MODE == 0) ? (tid & 63) : ((tid & 63) * 4);

    float              xr = 0.0f;       // MODE 0 staged value
    unsigned long long xq = 0ull;       // MODE 1 staged 4x bf16

    // preload t = 0
    {
        int tt = dir ? (T_STEPS - 1) : 0;
        if (MODE == 0) {
            if (sk < din) xr = xf[((size_t)(b0 + sb) * T_STEPS + tt) * din + sk];
        } else {
            xq = *(const unsigned long long*)(xb + ((size_t)(b0 + sb) * T_STEPS + tt) * 256 + sk);
        }
    }

    for (int t = 0; t < T_STEPS; ++t) {
        const int tt = dir ? (T_STEPS - 1 - t) : t;

        // commit staged x_t to A (pad region stays zero: xr==0 there)
        if (MODE == 0) {
            Ash[sb * KTOT + sk] = f2bf(xr);
        } else {
            *(unsigned long long*)&Ash[sb * KTOT + sk] = xq;
        }
        // issue x_{t+1} loads now; latency hidden behind WMMA + gate phases
        if (t + 1 < T_STEPS) {
            int tn = dir ? (T_STEPS - 2 - t) : (t + 1);
            if (MODE == 0) {
                if (sk < din) xr = xf[((size_t)(b0 + sb) * T_STEPS + tn) * din + sk];
            } else {
                xq = *(const unsigned long long*)(xb + ((size_t)(b0 + sb) * T_STEPS + tn) * 256 + sk);
            }
        }
        // warm L2 for t+2
        if (t + 2 < T_STEPS) {
            int tp = dir ? (T_STEPS - 3 - t) : (t + 2);
            if (MODE == 0)
                __builtin_prefetch(&xf[((size_t)(b0 + sb) * T_STEPS + tp) * din], 0, 1);
            else
                __builtin_prefetch(&xb[((size_t)(b0 + sb) * T_STEPS + tp) * 256 + sk], 0, 1);
        }
        __syncthreads();

        // ---- WMMA: this wave's 16x16 gate tile, K = KTOT, dual accumulators ----
        v8f acc0, acc1;
#pragma unroll
        for (int i = 0; i < 8; ++i) { acc0[i] = bias; acc1[i] = 0.0f; }
#pragma unroll
        for (int kf = 0; kf < KFRAGS; ++kf) {
            Frag af;
#pragma unroll
            for (int i = 0; i < 8; ++i) {
                // A: lanes0-15 hold K {0..7,16..23}, lanes16-31 {8..15,24..31}
                int k = 32 * kf + 8 * half + ((i < 4) ? (2 * i) : (2 * i + 8));
                af.u32[i] = *(const unsigned*)&Ash[ln * KTOT + k];
            }
            if (kf & 1)
                acc1 = __builtin_amdgcn_wmma_f32_16x16x32_bf16(
                    false, af.v, false, bfrag[kf].v, (short)0, acc1, false, false);
            else
                acc0 = __builtin_amdgcn_wmma_f32_16x16x32_bf16(
                    false, af.v, false, bfrag[kf].v, (short)0, acc0, false, false);
        }
#pragma unroll
        for (int i = 0; i < 8; ++i)
            Hg[(8 * half + i) * 512 + wv * 16 + ln] = acc0[i] + acc1[i];

        __syncthreads();

        // ---- gate nonlinearity + h update: 2 (b,u) elements per thread ----
#pragma unroll
        for (int j = 0; j < 2; ++j) {
            int e = tid + j * 1024;
            int b = e >> 7, u = e & 127;
            float r  = sigmoidf_(Hg[b * 512 + u]);
            float z  = sigmoidf_(Hg[b * 512 + 128 + u]);
            float xn = Hg[b * 512 + 256 + u];
            float hn = Hg[b * 512 + 384 + u];
            float n  = tanhf_(xn + r * hn);
            float hp = Hs[b * HDIM + u];
            float hw = (1.0f - z) * n + z * hp;
            Hs[b * HDIM + u]           = hw;
            Ash[b * KTOT + KXPAD + u]  = f2bf(hw);
            if (MODE == 0) {
                out_bf16[((size_t)(b0 + b) * T_STEPS + tt) * 256 + dir * 128 + u] = f2bf(hw);
            } else {
                if (j == 0) pacc0 += hw; else pacc1 += hw;
            }
        }
        // next iteration's staged write + loop-top barrier fence the A buffer
    }

    if (MODE == 1) {
#pragma unroll
        for (int j = 0; j < 2; ++j) {
            int e = tid + j * 1024;
            int b = e >> 7, u = e & 127;
            float s = (j == 0 ? pacc0 : pacc1) * (1.0f / T_STEPS);
            pooled[(size_t)(b0 + b) * 256 + dir * 128 + u] = s;
        }
    }
}

// FC: out[128,128] = pooled[128,256] @ fc_w^T + fc_b, one wave per 16x16 tile.
__global__ __launch_bounds__(32) void fc_kernel(
    const float* __restrict__ pooled, const float* __restrict__ fc_w,
    const float* __restrict__ fc_b, float* __restrict__ out)
{
    const int lane = threadIdx.x & 31;
    const int half = (lane >> 4) & 1;
    const int ln   = lane & 15;
    const int tm   = blockIdx.x >> 3;   // row tile (batch)
    const int tn   = blockIdx.x & 7;    // col tile (output unit)

    v8f acc;
    float bias = fc_b[tn * 16 + ln];
#pragma unroll
    for (int i = 0; i < 8; ++i) acc[i] = bias;

#pragma unroll
    for (int kf = 0; kf < 8; ++kf) {
        Frag af, bf;
#pragma unroll
        for (int i = 0; i < 8; ++i) {
            int ka = 32 * kf + 8 * half + ((i < 4) ? (2 * i) : (2 * i + 8));
            af.us[2 * i]     = f2bf(pooled[(size_t)(tm * 16 + ln) * 256 + ka]);
            af.us[2 * i + 1] = f2bf(pooled[(size_t)(tm * 16 + ln) * 256 + ka + 1]);
            int kb = 32 * kf + 8 * half + ((i < 4) ? (2 * i) : (2 * i + 8));
            bf.us[2 * i]     = f2bf(fc_w[(size_t)(tn * 16 + ln) * 256 + kb]);
            bf.us[2 * i + 1] = f2bf(fc_w[(size_t)(tn * 16 + ln) * 256 + kb + 1]);
        }
        acc = __builtin_amdgcn_wmma_f32_16x16x32_bf16(
            false, af.v, false, bf.v, (short)0, acc, false, false);
    }
#pragma unroll
    for (int i = 0; i < 8; ++i)
        out[(size_t)(tm * 16 + 8 * half + i) * 128 + tn * 16 + ln] = acc[i];
}

extern "C" void kernel_launch(void* const* d_in, const int* in_sizes, int n_in,
                              void* d_out, int out_size, void* d_ws, size_t ws_size,
                              hipStream_t stream)
{
    (void)in_sizes; (void)n_in; (void)out_size; (void)ws_size;
    const float* x     = (const float*)d_in[0];
    const float* w_ih0 = (const float*)d_in[1];
    const float* w_hh0 = (const float*)d_in[2];
    const float* b_ih0 = (const float*)d_in[3];
    const float* b_hh0 = (const float*)d_in[4];
    const float* w_ih1 = (const float*)d_in[5];
    const float* w_hh1 = (const float*)d_in[6];
    const float* b_ih1 = (const float*)d_in[7];
    const float* b_hh1 = (const float*)d_in[8];
    const float* fc_w  = (const float*)d_in[9];
    const float* fc_b  = (const float*)d_in[10];

    // workspace: out0 bf16 [128,1000,256] (65.5 MB, fits in L2), then pooled f32 [128,256]
    unsigned short* out0 = (unsigned short*)d_ws;
    float* pooled = (float*)((char*)d_ws +
                             (size_t)BATCH * T_STEPS * 256 * sizeof(unsigned short));

    // layer 0: Din=40 padded to 64, K = 64+128 = 192 (6 k-frags)
    bigru_scan<64, 192, 6, 0><<<dim3(16), dim3(1024), 0, stream>>>(
        (const void*)x, w_ih0, w_hh0, b_ih0, b_hh0, 40, out0, nullptr);
    // layer 1: Din=256, K = 256+128 = 384 (12 k-frags), accumulates time-mean
    bigru_scan<256, 384, 12, 1><<<dim3(16), dim3(1024), 0, stream>>>(
        (const void*)out0, w_ih1, w_hh1, b_ih1, b_hh1, 256, nullptr, pooled);
    // FC head
    fc_kernel<<<dim3(64), dim3(32), 0, stream>>>(pooled, fc_w, fc_b, (float*)d_out);
}
